// VectorQuantizer_32847909879838
// MI455X (gfx1250) — matrix operand order repaired and checked by reading, not compiled, not run on hardware
//
#include <hip/hip_runtime.h>

// ---------------------------------------------------------------------------
// MI455X (gfx1250) vector-quantizer.
//   score(m,k) = 64 + 0.5*|c_k|^2 - x_m . c_k    (same argmin as squared L2;
//                                                 +64 bias keeps score > 0)
// Folded into a single f16 WMMA GEMM with K padded 48 -> 64:
//   A row m = [-x(48), 1.0, 0 x15]   B col k = [c(48), 64+0.5|c|^2, 0 x15]
// Since scores are positive, their IEEE f32 bit patterns are monotonic as
// u32: argmin is carried by vector v_min_u32 on (bits & ~1023) | col.
// Codebook lives packed in LDS (128KB of the 320KB/WGP) in the exact
// v_wmma_f32_16x16x32_f16 B-fragment lane image.
// ---------------------------------------------------------------------------

typedef _Float16 half_t;
typedef __attribute__((ext_vector_type(16))) _Float16 v16h;
typedef __attribute__((ext_vector_type(8)))  _Float16 v8h;
typedef __attribute__((ext_vector_type(8)))  float    v8f;
typedef __attribute__((ext_vector_type(8)))  unsigned v8u;

#define IMG_H      2048
#define IMG_W      2048
#define CH         3
#define ROWPITCH   (IMG_W * CH)      // 6144 floats per image row
#define K_CODES    1024
#define D          48                // 16 pixels * 3 channels
#define KP         64                // padded reduction dim
#define NT         64                // 1024 / 16 codebook column tiles
#define WPB        4                 // waves per block
#define BT         (WPB * 32)        // 128 threads
#define NUM_MPAIRS 8192              // (2048/4 block-rows=512) * (512/32 chunks=16)
#define LDS_B_BYTES   (NT * 1024 * 2)            // 131072
#define LDS_A_BYTES   (WPB * 32 * KP * 2)        // 16384
#define LDS_IDX_BYTES (WPB * 32 * 4)             // 512
#define SMEM_BYTES    (LDS_B_BYTES + LDS_A_BYTES + LDS_IDX_BYTES)

// ---- Pack codebook into per-lane B-fragment image + folded biased norm ----
// B fragment layout (16-bit, 32x16 KxN per v16h): lane l holds column (l&15),
// halves s<8 = K[(l>>4)*8 + s], s>=8 = K[16 + (l>>4)*8 + s-8].  Fragment f covers K in [f*32, f*32+32).
// packed[t*1024 + f*512 + l*16 + s]
__global__ void vq_pack_codebook(const float* __restrict__ cb,
                                 half_t* __restrict__ pb) {
  int id = blockIdx.x * blockDim.x + threadIdx.x;   // 65536 total
  int t  = id >> 10;
  int r  = id & 1023;
  int f  = r >> 9;           // fragment 0/1
  int l  = (r >> 4) & 31;    // lane
  int s  = r & 15;           // half slot within fragment
  int n  = l & 15;           // column within tile
  int kh = l >> 4;
  int k  = f * 32 + (s < 8 ? kh * 8 + s : 16 + kh * 8 + (s - 8));
  int code = t * 16 + n;
  float v = 0.0f;
  if (k < D) {
    v = cb[code * D + k];
  } else if (k == D) {
    float acc = 0.0f;
    for (int q = 0; q < D; ++q) { float c = cb[code * D + q]; acc += c * c; }
    v = 0.5f * acc + 64.0f;   // bias guarantees positive scores
  }
  pb[t * 1024 + f * 512 + l * 16 + s] = (half_t)v;
}

// ---- Main kernel: 1 wave = 32 image blocks vs all 1024 codes --------------
__global__ __launch_bounds__(BT) void vq_main(const float* __restrict__ img,
                                              const float* __restrict__ cb,
                                              const half_t* __restrict__ pb,
                                              float* __restrict__ out) {
  extern __shared__ char smem[];
  half_t* ldsB    = (half_t*)smem;                               // 64 tiles
  half_t* stageA  = (half_t*)(smem + LDS_B_BYTES);               // per wave
  int*    codeIdx = (int*)   (smem + LDS_B_BYTES + LDS_A_BYTES); // per wave

  const int tid  = threadIdx.x;
  const int lane = tid & 31;
  const int wid  = tid >> 5;

  const int p   = blockIdx.x * WPB + wid;   // M-pair id, exactly 8192 waves
  const int by  = p >> 4;                   // block-row 0..511
  const int bx0 = (p & 15) * 32;            // first of 32 block-cols
  const int rowBase = (4 * by) * ROWPITCH + bx0 * 12;

  // Kick image lines into flight (global_prefetch_b8) before the LDS copy.
  #pragma unroll
  for (int i = 0; i < 4; ++i)
    __builtin_prefetch(img + rowBase + i * ROWPITCH + lane * 12, 0, 0);

  // Cooperative load of the whole packed codebook into LDS (128KB, once).
  {
    const uint4* src = (const uint4*)pb;
    uint4*       dst = (uint4*)ldsB;
    #pragma unroll 4
    for (int i = tid; i < LDS_B_BYTES / 16; i += BT) dst[i] = src[i];
  }
  __syncthreads();

  half_t* sA   = stageA  + wid * (32 * KP);
  int*    cIdx = codeIdx + wid * 32;

  // K-padding: k=48 -> 1.0 (picks up the folded biased norm), 49..63 -> 0.
  {
    int m = lane;
    sA[m * KP + 48] = (half_t)1.0f;
    #pragma unroll
    for (int q = 49; q < 64; ++q) sA[m * KP + q] = (half_t)0.0f;
  }

  // Stage -x as f16: 4 image rows x 384 contiguous floats, fully coalesced.
  #pragma unroll
  for (int i = 0; i < 4; ++i) {
    const float* rp = img + rowBase + i * ROWPITCH;
    #pragma unroll
    for (int t = 0; t < 12; ++t) {
      int   off = t * 32 + lane;          // 0..383
      float v   = rp[off];
      int   m   = off / 12;               // block within chunk
      int   e2  = off - m * 12;           // (j,c) within block row
      sA[m * KP + i * 12 + e2] = (half_t)(-v);
    }
  }

  // A fragments (16-bit A 16x32 lane image):
  // lane<16: row=lane,   halves 0..7 = K[kb..kb+7], 8..15 = K[kb+16..]
  // lane>=16: same rows, K bases shifted by 8 (kh=1).
  const int kh   = lane >> 4;
  const int colb = lane & 15;
  const half_t* p0 = sA + colb * KP;          // rows 0..15  (tile 0)
  const half_t* p1 = sA + (16 + colb) * KP;   // rows 16..31 (tile 1)
  #define LD8(P, O) (*(const v8h*)((P) + (O)))
  v16h a00 = __builtin_shufflevector(LD8(p0, kh * 8),      LD8(p0, 16 + kh * 8),
                                     0,1,2,3,4,5,6,7,8,9,10,11,12,13,14,15);
  v16h a01 = __builtin_shufflevector(LD8(p0, 32 + kh * 8), LD8(p0, 48 + kh * 8),
                                     0,1,2,3,4,5,6,7,8,9,10,11,12,13,14,15);
  v16h a10 = __builtin_shufflevector(LD8(p1, kh * 8),      LD8(p1, 16 + kh * 8),
                                     0,1,2,3,4,5,6,7,8,9,10,11,12,13,14,15);
  v16h a11 = __builtin_shufflevector(LD8(p1, 32 + kh * 8), LD8(p1, 48 + kh * 8),
                                     0,1,2,3,4,5,6,7,8,9,10,11,12,13,14,15);

  // Running (score,argmin) as sortable u32 keys, kept as 8-wide vectors.
  v8u mink0 = (v8u)(0xFFFFFFFFu);
  v8u mink1 = (v8u)(0xFFFFFFFFu);

  // Sweep all 64 column tiles: 4 WMMAs per column tile (2 M-tiles reuse B).
  #pragma unroll 4
  for (int t = 0; t < NT; ++t) {
    const half_t* bt = ldsB + t * 1024;
    v16h b0 = *(const v16h*)(bt + lane * 16);        // K 0..31
    v16h b1 = *(const v16h*)(bt + 512 + lane * 16);  // K 32..63
    v8f acc0 = {}, acc1 = {};
    acc0 = __builtin_amdgcn_wmma_f32_16x16x32_f16(false, a00, false, b0,
                                                  (short)0, acc0, false, false);
    acc0 = __builtin_amdgcn_wmma_f32_16x16x32_f16(false, a01, false, b1,
                                                  (short)0, acc0, false, false);
    acc1 = __builtin_amdgcn_wmma_f32_16x16x32_f16(false, a10, false, b0,
                                                  (short)0, acc1, false, false);
    acc1 = __builtin_amdgcn_wmma_f32_16x16x32_f16(false, a11, false, b1,
                                                  (short)0, acc1, false, false);
    unsigned cand = (unsigned)((t << 4) + colb);  // global codebook column
    // Vector key build + vector umin: v_and_or_b32 + v_min_u32 per element.
    v8u key0 = (__builtin_bit_cast(v8u, acc0) & 0xFFFFFC00u) | cand;
    v8u key1 = (__builtin_bit_cast(v8u, acc1) & 0xFFFFFC00u) | cand;
    mink0 = __builtin_elementwise_min(mink0, key0);
    mink1 = __builtin_elementwise_min(mink1, key1);
  }

  // Per-row argmin: C layout puts row d (lanes 0-15) / row 8+d (lanes 16-31)
  // in slot d -> 16-lane xor butterfly of v_min_u32 per slot.
  #pragma unroll
  for (int d = 0; d < 8; ++d) {
    unsigned k0 = mink0[d], k1 = mink1[d];
    #pragma unroll
    for (int m = 8; m >= 1; m >>= 1) {
      unsigned o0 = (unsigned)__shfl_xor((int)k0, m, 32);
      unsigned o1 = (unsigned)__shfl_xor((int)k1, m, 32);
      k0 = k0 < o0 ? k0 : o0;
      k1 = k1 < o1 ? k1 : o1;
    }
    if ((lane & 15) == 0) {
      int ro = (lane >> 4) * 8;          // lane 0 -> rows d, lane 16 -> 8+d
      cIdx[ro + d]      = (int)(k0 & 1023u);   // blocks 0..15
      cIdx[16 + ro + d] = (int)(k1 & 1023u);   // blocks 16..31
    }
  }
  // LDS ops are in-order within a wave: reads below see the writes above.

  // Gather winning codebook rows (fp32, L2-resident) -> coalesced row stores.
  #pragma unroll
  for (int i = 0; i < 4; ++i) {
    float* op = out + rowBase + i * ROWPITCH;
    #pragma unroll
    for (int t = 0; t < 12; ++t) {
      int off  = t * 32 + lane;
      int m    = off / 12;
      int e2   = off - m * 12;
      int code = cIdx[m];
      op[off] = cb[code * D + i * 12 + e2];
    }
  }
}

extern "C" void kernel_launch(void* const* d_in, const int* in_sizes, int n_in,
                              void* d_out, int out_size, void* d_ws, size_t ws_size,
                              hipStream_t stream) {
  const float* img = (const float*)d_in[0];   // 2048*2048*3 f32
  const float* cb  = (const float*)d_in[1];   // 1024*16*3  f32
  half_t*      pb  = (half_t*)d_ws;           // 128KB packed B fragments
  float*       out = (float*)d_out;

  vq_pack_codebook<<<dim3(256), dim3(256), 0, stream>>>(cb, pb);
  vq_main<<<dim3(NUM_MPAIRS / WPB), dim3(BT), SMEM_BYTES, stream>>>(img, cb, pb, out);
}
